// SimpleAttention_56453050138658
// MI455X (gfx1250) — matrix operand order repaired and checked by reading, not compiled, not run on hardware
//
#include <hip/hip_runtime.h>
#include <hip/hip_bf16.h>

// SimpleAttention for MI455X (gfx1250), wave32 + WMMA bf16 + async global->LDS.
//
// Kernel 1 (qgemm): qwqAll = concat(query,last)(4096x512) @ Wq(512x256) via WMMA.
// Kernel 2 (main, 1 block = 1 batch):
//   1. mem tile staged f32 into LDS via global_load_async_to_lds_b128 (ASYNCcnt),
//      overlapped with VALU Wh^T bf16 transpose staging
//   2. hidden_pre = mem @ Wh via v_wmma_f32_16x16x32_bf16 (A converted once/wave)
//   3. s[m] = sum_n tanh(.)*v[n] fused on C fragments (v_tanh_f32)
//   4. masked softmax, pooled = s^T @ mem from the f32 LDS tile (full precision)
// memory[] is read from HBM exactly once (~23 us roofline at 23.3 TB/s).

typedef __attribute__((ext_vector_type(16))) __bf16 v16bf;
typedef __attribute__((ext_vector_type(8)))  float  v8f;

#define NB  4096
#define TT  128
#define TM  127
#define DD  256
#define QD  512
#define F32STRIDE 260    // 256 + 4 f32 pad: 16B-aligned rows, stride%64 = 4 banks
#define LDSTRIDE  264    // 256 + 8 halves pad for bf16 Wh^T
#define QSTRIDE   520    // 512 + 8 halves pad

__device__ __forceinline__ unsigned short f32_to_bf16(float f) {
    unsigned int u = __float_as_uint(f);
    unsigned int r = (u + 0x7FFFu + ((u >> 16) & 1u)) >> 16;   // RNE
    return (unsigned short)r;
}
__device__ __forceinline__ unsigned int pack_bf16(float lo, float hi) {
    return ((unsigned int)f32_to_bf16(hi) << 16) | f32_to_bf16(lo);
}
__device__ __forceinline__ float fast_tanh(float x) {
#if __has_builtin(__builtin_amdgcn_tanhf)
    return __builtin_amdgcn_tanhf(x);            // v_tanh_f32 (CDNA5 trans op)
#elif __has_builtin(__builtin_amdgcn_tanh_f32)
    return __builtin_amdgcn_tanh_f32(x);
#else
    float e = __expf(2.0f * x);                  // branchless fallback
    return 1.0f - 2.0f / (e + 1.0f);
#endif
}

union Frag { v16bf v; uint4 u[2]; };

// ---------------------------------------------------------------------------
// Kernel 1: qwqAll[b][n] = sum_k concat(query,last)[b][k] * Wq[k][n]
// 256 blocks x 256 threads; block handles 16 batches (M=16, N=256, K=512).
// ---------------------------------------------------------------------------
__global__ void __launch_bounds__(256)
qgemm_kernel(const float* __restrict__ memory,
             const float* __restrict__ query,
             const float* __restrict__ Wq,
             float* __restrict__ qwqAll)
{
    extern __shared__ char smem[];
    unsigned short* ldsWqT = (unsigned short*)smem;          // 256 x 520 bf16 : Wq^T[n][k]
    unsigned short* ldsA   = ldsWqT + 256 * QSTRIDE;         // 16  x 520 bf16

    const int tid  = threadIdx.x;
    const int lane = tid & 31;
    const int wave = tid >> 5;
    const int hi   = lane >> 4;
    const int nLow = lane & 15;
    const int b0   = blockIdx.x * 16;

    // stage Wq^T (global reads coalesced over n=tid; packed b32 LDS stores)
    for (int k2 = 0; k2 < 256; ++k2) {
        float a = Wq[(2 * k2 + 0) * DD + tid];
        float b = Wq[(2 * k2 + 1) * DD + tid];
        *(unsigned int*)(ldsWqT + tid * QSTRIDE + 2 * k2) = pack_bf16(a, b);
    }

    // stage A: 16 rows of concat(query[b], memory[b,127,:]), 512 wide
    for (int i = 0; i < 32; ++i) {
        int idx = i * 256 + tid;
        int m   = idx >> 9;
        int k   = idx & 511;
        int bb  = b0 + m;
        float f = (k < DD) ? query[(size_t)bb * DD + k]
                           : memory[(size_t)bb * TT * DD + (size_t)TM * DD + (k - DD)];
        ldsA[m * QSTRIDE + k] = f32_to_bf16(f);
    }
    __syncthreads();

    // hoist all 16 A fragments (shared by both N-tiles of this wave)
    Frag AF[16];
    {
        const unsigned short* arow = ldsA + nLow * QSTRIDE;
        #pragma unroll
        for (int kt = 0; kt < 16; ++kt) {
            int k0 = kt * 32 + hi * 8;
            AF[kt].u[0] = *(const uint4*)(arow + k0);
            AF[kt].u[1] = *(const uint4*)(arow + k0 + 16);
        }
    }

    for (int t = 0; t < 2; ++t) {
        int nt = wave * 2 + t;
        v8f c;
        #pragma unroll
        for (int j = 0; j < 8; ++j) c[j] = 0.0f;
        const unsigned short* wrow = ldsWqT + (nt * 16 + nLow) * QSTRIDE + hi * 8;
        #pragma unroll
        for (int kt = 0; kt < 16; ++kt) {
            Frag Bf;
            Bf.u[0] = *(const uint4*)(wrow + kt * 32);
            Bf.u[1] = *(const uint4*)(wrow + kt * 32 + 16);
            c = __builtin_amdgcn_wmma_f32_16x16x32_bf16(
                    false, AF[kt].v, false, Bf.v, (short)0, c, false, false);
        }
        #pragma unroll
        for (int j = 0; j < 8; ++j) {
            int m = j + 8 * hi;           // C layout: c[j] = C[m][n], n = nLow
            qwqAll[(size_t)(b0 + m) * DD + nt * 16 + nLow] = c[j];
        }
    }
}

// ---------------------------------------------------------------------------
// Kernel 2: main attention, 4096 blocks x 256 threads (8 waves), 1 batch/block.
// ---------------------------------------------------------------------------
__global__ void __launch_bounds__(256)
simple_attention_kernel(const float* __restrict__ memory,
                        const unsigned char* __restrict__ mask,   // jax bool -> 1 byte
                        const float* __restrict__ Wh,
                        const float* __restrict__ vvec,
                        const float* __restrict__ qwqAll,
                        float* __restrict__ out)
{
    extern __shared__ char smem[];
    float*          ldsMemF = (float*)smem;                      // 128 x 260 f32 (row 127 = 0)
    unsigned short* ldsWh   = (unsigned short*)(ldsMemF + 128 * F32STRIDE); // 256 x 264 bf16
    float* sArr = (float*)(ldsWh + 256 * LDSTRIDE);              // 128 f32
    float* wArr = sArr + 128;                                    // 128 f32

    const int b    = blockIdx.x;
    const int tid  = threadIdx.x;
    const int lane = tid & 31;
    const int wave = tid >> 5;
    const int hi   = lane >> 4;
    const int nLow = lane & 15;

    const float* memB = memory + (size_t)b * TT * DD;

    // ---- issue async global->LDS copies for the mem tile FIRST (overlaps below) ----
    // dynamic LDS starts at offset 0 of the wave's LDS base; b128 per lane.
    {
        int sub = tid >> 6;               // 0..3
        int col = (tid & 63) * 4;         // 0..252
        for (int r4 = 0; r4 < 32; ++r4) {
            int row = r4 * 4 + sub;
            if (row < TM) {
                unsigned int ldsOff = (unsigned int)((row * F32STRIDE + col) * 4);
                unsigned long long ga = (unsigned long long)(memB + row * DD + col);
                asm volatile("global_load_async_to_lds_b128 %0, %1, off"
                             :: "v"(ldsOff), "v"(ga) : "memory");
            }
        }
    }

    // ---- preload loop-invariant per-lane scalars ----
    float vReg[16], qwqReg[16];
    #pragma unroll
    for (int nt = 0; nt < 16; ++nt) {
        vReg[nt]   = vvec[nt * 16 + nLow];
        qwqReg[nt] = qwqAll[(size_t)b * DD + nt * 16 + nLow];
    }

    // ---- stage Wh^T -> LDS bf16 (float4 loads, transposed b16 stores) ----
    {
        int sub = tid >> 6;
        int n4  = (tid & 63) * 4;
        for (int i = 0; i < 64; ++i) {
            int k = i * 4 + sub;
            float4 w4 = *(const float4*)(Wh + k * DD + n4);
            ldsWh[(n4 + 0) * LDSTRIDE + k] = f32_to_bf16(w4.x);
            ldsWh[(n4 + 1) * LDSTRIDE + k] = f32_to_bf16(w4.y);
            ldsWh[(n4 + 2) * LDSTRIDE + k] = f32_to_bf16(w4.z);
            ldsWh[(n4 + 3) * LDSTRIDE + k] = f32_to_bf16(w4.w);
        }
    }

    // ---- zero row 127 (read by wave 7's A fragments; excluded from pooling) ----
    for (int i = tid; i < F32STRIDE; i += 256) ldsMemF[TM * F32STRIDE + i] = 0.0f;

    asm volatile("s_wait_asynccnt 0" ::: "memory");   // own async copies done
    __syncthreads();                                   // whole tile visible to WG

    // ---- build this wave's A strip (16 rows x K=256) as bf16 fragments ----
    Frag A[8];
    {
        const float* rowp = ldsMemF + (wave * 16 + nLow) * F32STRIDE;
        #pragma unroll
        for (int kt = 0; kt < 8; ++kt) {
            int k0 = kt * 32 + hi * 8;
            float4 f0 = *(const float4*)(rowp + k0);
            float4 f1 = *(const float4*)(rowp + k0 + 4);
            float4 f2 = *(const float4*)(rowp + k0 + 16);
            float4 f3 = *(const float4*)(rowp + k0 + 20);
            A[kt].u[0] = make_uint4(pack_bf16(f0.x, f0.y), pack_bf16(f0.z, f0.w),
                                    pack_bf16(f1.x, f1.y), pack_bf16(f1.z, f1.w));
            A[kt].u[1] = make_uint4(pack_bf16(f2.x, f2.y), pack_bf16(f2.z, f2.w),
                                    pack_bf16(f3.x, f3.y), pack_bf16(f3.z, f3.w));
        }
    }

    // ---- GEMM + fused tanh + dot(v): 16 N-tiles x 8 WMMA K-steps ----
    float sAcc[8];
    #pragma unroll
    for (int j = 0; j < 8; ++j) sAcc[j] = 0.0f;

    for (int nt = 0; nt < 16; ++nt) {
        v8f c;
        float cinit = qwqReg[nt];
        #pragma unroll
        for (int j = 0; j < 8; ++j) c[j] = cinit;

        const unsigned short* wrow = ldsWh + (nt * 16 + nLow) * LDSTRIDE + hi * 8;
        #pragma unroll
        for (int kt = 0; kt < 8; ++kt) {
            Frag Bf;
            Bf.u[0] = *(const uint4*)(wrow + kt * 32);
            Bf.u[1] = *(const uint4*)(wrow + kt * 32 + 16);
            c = __builtin_amdgcn_wmma_f32_16x16x32_bf16(
                    false, A[kt].v, false, Bf.v, (short)0, c, false, false);
        }

        float vn = vReg[nt];
        #pragma unroll
        for (int j = 0; j < 8; ++j)
            sAcc[j] = fmaf(fast_tanh(c[j]), vn, sAcc[j]);
    }

    // ---- reduce scores across the 16 lanes holding each row ----
    #pragma unroll
    for (int j = 0; j < 8; ++j) {
        float t = sAcc[j];
        t += __shfl_xor(t, 1, 32);
        t += __shfl_xor(t, 2, 32);
        t += __shfl_xor(t, 4, 32);
        t += __shfl_xor(t, 8, 32);
        if (nLow == 0) sArr[wave * 16 + hi * 8 + j] = t;
    }
    __syncthreads();

    // ---- softmax over 127 scores, mask, renormalize (wave 0) ----
    if (wave == 0) {
        float sv[4], ev[4], pm[4];
        float mx = -1e30f;
        #pragma unroll
        for (int i = 0; i < 4; ++i) {
            int t = lane + 32 * i;
            sv[i] = (t < TM) ? sArr[t] : -1e30f;
            mx = fmaxf(mx, sv[i]);
        }
        for (int off = 16; off; off >>= 1) mx = fmaxf(mx, __shfl_xor(mx, off, 32));
        float se = 0.0f;
        #pragma unroll
        for (int i = 0; i < 4; ++i) {
            ev[i] = (sv[i] > -1e29f) ? __expf(sv[i] - mx) : 0.0f;
            se += ev[i];
        }
        for (int off = 16; off; off >>= 1) se += __shfl_xor(se, off, 32);
        float sm = 0.0f;
        #pragma unroll
        for (int i = 0; i < 4; ++i) {
            int t = lane + 32 * i;
            float mval = (t < TM) ? (float)mask[(size_t)b * TT + t] : 0.0f;
            pm[i] = (ev[i] / se) * mval;
            sm += pm[i];
        }
        for (int off = 16; off; off >>= 1) sm += __shfl_xor(sm, off, 32);
        float inv = 1.0f / (sm + 1e-7f);
        #pragma unroll
        for (int i = 0; i < 4; ++i) {
            int t = lane + 32 * i;
            if (t < 128) wArr[t] = pm[i] * inv;
        }
    }
    __syncthreads();

    // ---- pooled[d] = sum_t w[t] * mem[t][d] from the f32 LDS tile ----
    {
        float acc = 0.0f;
        #pragma unroll 4
        for (int t = 0; t < TM; ++t)
            acc = fmaf(wArr[t], ldsMemF[t * F32STRIDE + tid], acc);
        out[(size_t)b * QD + tid]      = acc;
        out[(size_t)b * QD + DD + tid] = memB[(size_t)TM * DD + tid];   // last
    }
}

extern "C" void kernel_launch(void* const* d_in, const int* in_sizes, int n_in,
                              void* d_out, int out_size, void* d_ws, size_t ws_size,
                              hipStream_t stream) {
    (void)in_sizes; (void)n_in; (void)out_size; (void)ws_size;
    const float*         memory = (const float*)d_in[0];
    const float*         query  = (const float*)d_in[1];
    const unsigned char* mask   = (const unsigned char*)d_in[2];
    const float*         Wq     = (const float*)d_in[3];
    const float*         Wh     = (const float*)d_in[4];
    const float*         v      = (const float*)d_in[5];
    float*               out    = (float*)d_out;
    float*               qwqAll = (float*)d_ws;            // 4096*256*4 = 4 MB scratch

    size_t shmem_q = (size_t)(256 + 16) * QSTRIDE * sizeof(unsigned short);   // ~276 KB
    size_t shmem_m = (size_t)128 * F32STRIDE * sizeof(float)
                   + (size_t)256 * LDSTRIDE * sizeof(unsigned short)
                   + 256 * sizeof(float);                                     // ~263 KB

    (void)hipFuncSetAttribute(reinterpret_cast<const void*>(qgemm_kernel),
                              hipFuncAttributeMaxDynamicSharedMemorySize, (int)shmem_q);
    (void)hipFuncSetAttribute(reinterpret_cast<const void*>(simple_attention_kernel),
                              hipFuncAttributeMaxDynamicSharedMemorySize, (int)shmem_m);

    qgemm_kernel<<<NB / 16, 256, shmem_q, stream>>>(memory, query, Wq, qwqAll);
    simple_attention_kernel<<<NB, 256, shmem_m, stream>>>(memory, mask, Wh, v, qwqAll, out);
}
